// TASA_attention_5325759447457
// MI455X (gfx1250) — matrix operand order repaired and checked by reading, not compiled, not run on hardware
//
#include <hip/hip_runtime.h>
#include <math.h>

// ---------------- problem constants ----------------
constexpr int kB  = 4;
constexpr int kT  = 1024;
constexpr int kDM = 512;
constexpr int kH  = 8;
constexpr int kDK = 64;

typedef __bf16 bf16;
typedef __bf16 v8bf  __attribute__((ext_vector_type(8)));
typedef __bf16 v16bf __attribute__((ext_vector_type(16)));
typedef float  v8f   __attribute__((ext_vector_type(8)));
typedef float  v8ff  __attribute__((ext_vector_type(8)));

typedef unsigned int u32x4 __attribute__((ext_vector_type(4)));
typedef int          i32x4 __attribute__((ext_vector_type(4)));
typedef int          i32x8 __attribute__((ext_vector_type(8)));

static __device__ __forceinline__ v8f wmma_bf16(v16bf a, v16bf b, v8f c) {
  // V_WMMA_F32_16X16X32_BF16 : D = A(16x32) * B(32x16) + C(16x16), f32 accum
  return __builtin_amdgcn_wmma_f32_16x16x32_bf16(false, a, false, b, (short)0, c,
                                                 false, false);
}

// A fragment (16x32, row-major source): lane m=lane&15 is the row,
// lanes 0-15 hold K = {0..7, 16..23}, lanes 16-31 hold K = {8..15, 24..31}.
static __device__ __forceinline__ v16bf ldfragA(const bf16* rowbase, int kk, int kgrp) {
  const v8bf lo = *(const v8bf*)(rowbase + kk + kgrp * 8);
  const v8bf hi = *(const v8bf*)(rowbase + kk + 16 + kgrp * 8);
  return __builtin_shufflevector(lo, hi, 0,1,2,3,4,5,6,7,8,9,10,11,12,13,14,15);
}
// B fragment (32x16): lane n=lane&15 is the column, lanes 0-15 hold K = {0..7, 16..23},
// lanes 16-31 hold K = {8..15, 24..31}; source row (per output col) is contiguous in K.
static __device__ __forceinline__ v16bf ldfragB(const bf16* colbase, int kk, int kgrp) {
  const v8bf lo = *(const v8bf*)(colbase + kk + kgrp * 8);
  const v8bf hi = *(const v8bf*)(colbase + kk + 16 + kgrp * 8);
  return __builtin_shufflevector(lo, hi, 0,1,2,3,4,5,6,7,8,9,10,11,12,13,14,15);
}
// A fragment built from an f32 source (converted on the fly).
static __device__ __forceinline__ v16bf cvtfragA(const float* rowbase, int kk, int kgrp) {
  v8ff lo = *(const v8ff*)(rowbase + kk + kgrp * 8);
  v8ff hi = *(const v8ff*)(rowbase + kk + 16 + kgrp * 8);
  v16bf a;
#pragma unroll
  for (int i = 0; i < 8; ++i) { a[i] = (bf16)lo[i]; a[i + 8] = (bf16)hi[i]; }
  return a;
}

// ---------------- TDM: stage [8ch][3rows][1024] tile into LDS ----------------
// D# per CDNA5 ISA ch.8: group0 = {count/lds_addr/global_addr/type},
// group1 = dims/strides, group2 = tensor_dim2, group3 = 0.
// dsz: 1 => 2-byte elements, 2 => 4-byte elements (dims are in elements).
static __device__ __forceinline__ void tdm_load_3rows(unsigned lds_byte,
                                                      const void* gsrc,
                                                      unsigned dsz) {
  const unsigned long long ga = (unsigned long long)(size_t)gsrc;
  u32x4 g0 = { 1u,                                    // count=1, user mode
               lds_byte,                              // LDS byte address
               (unsigned)(ga & 0xffffffffu),          // global_addr[31:0]
               (unsigned)((ga >> 32) & 0x01ffffffu) | (2u << 30) }; // addr[56:32] | type=2
  i32x8 g1 = { (int)(dsz << 16),       // data_size
               (int)(1024u << 16),     // tensor_dim0 = 1024
               (int)(3u << 16),        // tensor_dim1 = 3
               (int)(1024u << 16),     // tile_dim0 = 1024
               (int)((8u << 16) | 3u), // tile_dim2 = 8 | tile_dim1 = 3
               1024,                   // tensor_dim0_stride = 1024
               0,
               16 };                   // tensor_dim1_stride = 1024*1024 (>>16)
  i32x4 g2 = { 8, 0, 0, 0 };           // tensor_dim2 = 8
  i32x4 g3 = { 0, 0, 0, 0 };
  i32x8 g4 = { 0, 0, 0, 0, 0, 0, 0, 0 };  // clang-23 form: extra (unused) group
  __builtin_amdgcn_tensor_load_to_lds(g0, g1, g2, g3, g4, 0);
}

// ============================================================
// Kernel 0: f32 -> bf16 conversion (inputs + weights), grid-stride
// ============================================================
__global__ __launch_bounds__(256)
void k_cvt(const float* __restrict__ src, bf16* __restrict__ dst, int n) {
  for (int i = blockIdx.x * 256 + threadIdx.x; i < n; i += gridDim.x * 256)
    dst[i] = (bf16)src[i];
}

// ============================================================
// Kernel 1: fused QKV projections (bf16 WMMA, K=32, pipelined).
// wave -> 16(t) x 64(head dims) tile.
// which==0 -> Q (b,h,t,d) ; 1 -> K (b,h,t,d) ; 2 -> V^T (b,h,d,t)
// ============================================================
__global__ __launch_bounds__(32)
void k_qkv(const bf16* __restrict__ xq, const bf16* __restrict__ xk,
           const bf16* __restrict__ xv,
           const bf16* __restrict__ wq, const bf16* __restrict__ wk,
           const bf16* __restrict__ wv,
           bf16* __restrict__ Qo, bf16* __restrict__ Ko, bf16* __restrict__ Vt) {
  const int tile = blockIdx.x, h = blockIdx.y, which = blockIdx.z;
  const int lane = threadIdx.x;
  const int m = lane & 15, kgrp = lane >> 4;

  const bf16* X = (which == 0) ? xq : (which == 1) ? xk : xv;
  const bf16* W = (which == 0) ? wq : (which == 1) ? wk : wv;

  const int bt0 = tile * 16;
  const bf16* arow = X + (size_t)(bt0 + m) * kDM;
  const bf16* W0 = W + (size_t)(h * 64 +  0 + m) * kDM;
  const bf16* W1 = W + (size_t)(h * 64 + 16 + m) * kDM;
  const bf16* W2 = W + (size_t)(h * 64 + 32 + m) * kDM;
  const bf16* W3 = W + (size_t)(h * 64 + 48 + m) * kDM;

  v8f c0 = {}, c1 = {}, c2 = {}, c3 = {};
  v16bf a_n  = ldfragA(arow, 0, kgrp);
  v16bf b0_n = ldfragB(W0, 0, kgrp), b1_n = ldfragB(W1, 0, kgrp);
  v16bf b2_n = ldfragB(W2, 0, kgrp), b3_n = ldfragB(W3, 0, kgrp);
  for (int kk = 0; kk < kDM; kk += 32) {
    v16bf a = a_n, b0 = b0_n, b1 = b1_n, b2 = b2_n, b3 = b3_n;
    const int nk = kk + 32;
    if (nk < kDM) {            // prefetch next step before consuming current
      a_n  = ldfragA(arow, nk, kgrp);
      b0_n = ldfragB(W0, nk, kgrp); b1_n = ldfragB(W1, nk, kgrp);
      b2_n = ldfragB(W2, nk, kgrp); b3_n = ldfragB(W3, nk, kgrp);
    }
    c0 = wmma_bf16(a, b0, c0);
    c1 = wmma_bf16(a, b1, c1);
    c2 = wmma_bf16(a, b2, c2);
    c3 = wmma_bf16(a, b3, c3);
  }

  const int bIdx = bt0 / kT, t0 = bt0 % kT;
  v8f cc[4] = {c0, c1, c2, c3};
#pragma unroll
  for (int j = 0; j < 4; ++j) {
#pragma unroll
    for (int i = 0; i < 8; ++i) {
      const int t = t0 + i + 8 * kgrp;
      const int d = j * 16 + m;
      const bf16 val = (bf16)cc[j][i];
      if (which == 0)
        Qo[(((size_t)(bIdx * kH + h)) * kT + t) * kDK + d] = val;
      else if (which == 1)
        Ko[(((size_t)(bIdx * kH + h)) * kT + t) * kDK + d] = val;
      else
        Vt[(((size_t)(bIdx * kH + h)) * kDK + d) * kT + t] = val;
    }
  }
}

// ============================================================
// Kernel 2: M = Q K^T per (b,h) -> bf16.  Wave -> 16(q) x 64(k) tile.
// ============================================================
__global__ __launch_bounds__(32)
void k_scores(const bf16* __restrict__ Q, const bf16* __restrict__ K,
              bf16* __restrict__ M) {
  const int ktile = blockIdx.x, qtile = blockIdx.y, bh = blockIdx.z;
  const int lane = threadIdx.x;
  const int m = lane & 15, kgrp = lane >> 4;

  const bf16* Qb = Q + (size_t)bh * kT * kDK;
  const bf16* Kb = K + (size_t)bh * kT * kDK;
  const bf16* arow = Qb + (size_t)(qtile * 16 + m) * kDK;
  const bf16* K0 = Kb + (size_t)(ktile * 64 +  0 + m) * kDK;
  const bf16* K1 = Kb + (size_t)(ktile * 64 + 16 + m) * kDK;
  const bf16* K2 = Kb + (size_t)(ktile * 64 + 32 + m) * kDK;
  const bf16* K3 = Kb + (size_t)(ktile * 64 + 48 + m) * kDK;

  v8f c0 = {}, c1 = {}, c2 = {}, c3 = {};
#pragma unroll
  for (int kk = 0; kk < kDK; kk += 32) {
    v16bf a = ldfragA(arow, kk, kgrp);
    c0 = wmma_bf16(a, ldfragB(K0, kk, kgrp), c0);
    c1 = wmma_bf16(a, ldfragB(K1, kk, kgrp), c1);
    c2 = wmma_bf16(a, ldfragB(K2, kk, kgrp), c2);
    c3 = wmma_bf16(a, ldfragB(K3, kk, kgrp), c3);
  }

  bf16* Mb = M + (size_t)bh * kT * kT;
  v8f cc[4] = {c0, c1, c2, c3};
#pragma unroll
  for (int j = 0; j < 4; ++j)
#pragma unroll
    for (int i = 0; i < 8; ++i)
      Mb[(size_t)(qtile * 16 + i + 8 * kgrp) * kT + ktile * 64 + j * 16 + m] =
          (bf16)cc[j][i];
}

// ============================================================
// Kernel 3: Mt = conv3x3(prev, tw) + tb -> bf16.
// Block = (b,y); stages prev[8ch][3rows][1024] f32 into LDS via TDM
// (manual fallback on edge rows), then VALU conv.
// ============================================================
constexpr int kTconvLds = (kH * 3 * kT) * 4 + (kH * kH * 9) * 4;  // 100,608 B

__global__ __launch_bounds__(256)
void k_tconv(const float* __restrict__ prev, const float* __restrict__ tw,
             const float* __restrict__ tb, bf16* __restrict__ Mt) {
  extern __shared__ char smem[];
  float* in = (float*)smem;                    // [8][3][1024]
  float* w  = in + kH * 3 * kT;                // [8][8][3][3]
  const int tid = threadIdx.x;
  const int y = blockIdx.x, b = blockIdx.y;

  for (int i = tid; i < kH * kH * 9; i += 256) w[i] = tw[i];

  const bool edge = (y == 0) || (y == kT - 1);
  if (!edge) {
    if (tid < 32) {   // wave 0 issues the tensor DMA for all 8 planes x 3 rows
      const float* gsrc = prev + (((size_t)(b * kH)) * kT + (y - 1)) * kT;
      tdm_load_3rows((unsigned)(size_t)(void*)in, gsrc, 2u);
      __builtin_amdgcn_s_wait_tensorcnt(0);
    }
  } else {
    for (int idx = tid; idx < kH * 3 * kT; idx += 256) {
      const int x = idx & (kT - 1);
      const int r = (idx >> 10) % 3;
      const int c = idx / (3 * kT);
      const int yy = y + r - 1;
      in[idx] = (yy >= 0 && yy < kT)
                    ? prev[(((size_t)(b * kH + c)) * kT + yy) * kT + x]
                    : 0.0f;
    }
  }
  __syncthreads();

  const int x0 = tid * 4;
  float acc[kH][4];
#pragma unroll
  for (int h = 0; h < kH; ++h) {
    const float bias = tb[h];
#pragma unroll
    for (int p = 0; p < 4; ++p) acc[h][p] = bias;
  }

  for (int ci = 0; ci < kH; ++ci) {
#pragma unroll
    for (int r = 0; r < 3; ++r) {
      const float* rowp = in + (ci * 3 + r) * kT;
      float win[6];
#pragma unroll
      for (int u = 0; u < 6; ++u) {
        const int xx = x0 + u - 1;
        win[u] = (xx >= 0 && xx < kT) ? rowp[xx] : 0.0f;
      }
#pragma unroll
      for (int h = 0; h < kH; ++h) {
        const float* wp = &w[((h * kH + ci) * 3 + r) * 3];
        const float w0 = wp[0], w1 = wp[1], w2 = wp[2];
#pragma unroll
        for (int p = 0; p < 4; ++p)
          acc[h][p] = fmaf(w0, win[p],
                       fmaf(w1, win[p + 1],
                        fmaf(w2, win[p + 2], acc[h][p])));
      }
    }
  }
#pragma unroll
  for (int h = 0; h < kH; ++h)
#pragma unroll
    for (int p = 0; p < 4; ++p)
      Mt[(((size_t)(b * kH + h)) * kT + y) * kT + x0 + p] = (bf16)acc[h][p];
}

// ============================================================
// Kernel 4: A = softmax(mask(conv3x3(concat(M,Mt), aw)/8)) -> f32 (output)
// Block = (b,y); stages 16ch x 3rows x 1024 *bf16* (96KB) via two TDM loads.
// ============================================================
static __device__ __forceinline__ float blk_reduce_max(float v, float* red, int tid) {
#pragma unroll
  for (int o = 16; o > 0; o >>= 1) v = fmaxf(v, __shfl_xor(v, o, 32));
  if ((tid & 31) == 0) red[tid >> 5] = v;
  __syncthreads();
  if (tid < 8) {
    float t = red[tid];
#pragma unroll
    for (int o = 4; o > 0; o >>= 1) t = fmaxf(t, __shfl_xor(t, o, 8));
    if (tid == 0) red[0] = t;
  }
  __syncthreads();
  const float r = red[0];
  __syncthreads();
  return r;
}
static __device__ __forceinline__ float blk_reduce_sum(float v, float* red, int tid) {
#pragma unroll
  for (int o = 16; o > 0; o >>= 1) v += __shfl_xor(v, o, 32);
  if ((tid & 31) == 0) red[tid >> 5] = v;
  __syncthreads();
  if (tid < 8) {
    float t = red[tid];
#pragma unroll
    for (int o = 4; o > 0; o >>= 1) t += __shfl_xor(t, o, 8);
    if (tid == 0) red[0] = t;
  }
  __syncthreads();
  const float r = red[0];
  __syncthreads();
  return r;
}

constexpr int kAInElems = 16 * 3 * kT;                               // bf16
constexpr int kAWFloats = kH * 16 * 9;                               // 1152
constexpr int kAconvLds = kAInElems * 2 + kAWFloats * 4 + 32 * 4;    // 103,040 B

__global__ __launch_bounds__(256)
void k_aconv_softmax(const bf16* __restrict__ M, const bf16* __restrict__ Mt,
                     const float* __restrict__ aw, const float* __restrict__ ab,
                     const int* __restrict__ mask, float* __restrict__ A) {
  extern __shared__ char smem[];
  bf16*  in  = (bf16*)smem;                        // [16][3][1024]
  float* w   = (float*)(smem + kAInElems * 2);     // [8][16][9]
  float* red = w + kAWFloats;

  const int tid = threadIdx.x;
  const int y = blockIdx.x, b = blockIdx.y;

  for (int i = tid; i < kAWFloats; i += 256) w[i] = aw[i];

  const bool edge = (y == 0) || (y == kT - 1);
  if (!edge) {
    if (tid < 32) {
      const unsigned ldsb = (unsigned)(size_t)(void*)in;
      const bf16* gm  = M  + (((size_t)(b * kH)) * kT + (y - 1)) * kT;
      const bf16* gmt = Mt + (((size_t)(b * kH)) * kT + (y - 1)) * kT;
      tdm_load_3rows(ldsb, gm, 1u);                        // channels 0..7
      tdm_load_3rows(ldsb + kH * 3 * kT * 2, gmt, 1u);     // channels 8..15
      __builtin_amdgcn_s_wait_tensorcnt(0);
    }
  } else {
    for (int idx = tid; idx < kAInElems; idx += 256) {
      const int x = idx & (kT - 1);
      const int r = (idx >> 10) % 3;
      const int c = idx / (3 * kT);
      const int yy = y + r - 1;
      bf16 val = (bf16)0.0f;
      if (yy >= 0 && yy < kT)
        val = (c < 8) ? M [(((size_t)(b * kH + c))     * kT + yy) * kT + x]
                      : Mt[(((size_t)(b * kH + c - 8)) * kT + yy) * kT + x];
      in[idx] = val;
    }
  }
  __syncthreads();

  const int x0 = tid * 4;
  int mk[4];
#pragma unroll
  for (int p = 0; p < 4; ++p) mk[p] = mask[b * kT + x0 + p];

  float acc[kH][4];
#pragma unroll
  for (int h = 0; h < kH; ++h) {
    const float bias = ab[h];
#pragma unroll
    for (int p = 0; p < 4; ++p) acc[h][p] = bias;
  }

  for (int c = 0; c < 16; ++c) {
#pragma unroll
    for (int r = 0; r < 3; ++r) {
      const bf16* rowp = in + (c * 3 + r) * kT;
      float win[6];
#pragma unroll
      for (int u = 0; u < 6; ++u) {
        const int xx = x0 + u - 1;
        win[u] = (xx >= 0 && xx < kT) ? (float)rowp[xx] : 0.0f;
      }
#pragma unroll
      for (int h = 0; h < kH; ++h) {
        const float* wp = &w[((h * 16 + c) * 3 + r) * 3];
        const float w0 = wp[0], w1 = wp[1], w2 = wp[2];
#pragma unroll
        for (int p = 0; p < 4; ++p)
          acc[h][p] = fmaf(w0, win[p],
                       fmaf(w1, win[p + 1],
                        fmaf(w2, win[p + 2], acc[h][p])));
      }
    }
  }

  const float scale = 0.125f;  // 1/sqrt(64)
  for (int h = 0; h < kH; ++h) {
    float a4[4];
#pragma unroll
    for (int p = 0; p < 4; ++p)
      a4[p] = mk[p] ? acc[h][p] * scale : -1.0e9f;

    float lm = fmaxf(fmaxf(a4[0], a4[1]), fmaxf(a4[2], a4[3]));
    const float rmax = blk_reduce_max(lm, red, tid);

    float e[4], ls = 0.0f;
#pragma unroll
    for (int p = 0; p < 4; ++p) { e[p] = __expf(a4[p] - rmax); ls += e[p]; }
    const float rsum = blk_reduce_sum(ls, red, tid);
    const float inv = 1.0f / rsum;

    float* Ar = A + (((size_t)(b * kH + h)) * kT + y) * kT + x0;
#pragma unroll
    for (int p = 0; p < 4; ++p) Ar[p] = e[p] * inv;
  }
}

// ============================================================
// Kernel 5: Ctx = A * V per (b,h) -> bf16.  A is f32 (converted inline).
// ============================================================
__global__ __launch_bounds__(32)
void k_context(const float* __restrict__ A, const bf16* __restrict__ Vt,
               bf16* __restrict__ Ctx) {
  const int qtile = blockIdx.x, bh = blockIdx.y;
  const int lane = threadIdx.x;
  const int m = lane & 15, kgrp = lane >> 4;

  const float* Ab = A  + (size_t)bh * kT * kT;
  const bf16*  Vb = Vt + (size_t)bh * kDK * kT;
  const float* arow = Ab + (size_t)(qtile * 16 + m) * kT;
  const bf16* V0 = Vb + (size_t)( 0 + m) * kT;
  const bf16* V1 = Vb + (size_t)(16 + m) * kT;
  const bf16* V2 = Vb + (size_t)(32 + m) * kT;
  const bf16* V3 = Vb + (size_t)(48 + m) * kT;

  v8f c0 = {}, c1 = {}, c2 = {}, c3 = {};
  v16bf a_n  = cvtfragA(arow, 0, kgrp);
  v16bf b0_n = ldfragB(V0, 0, kgrp), b1_n = ldfragB(V1, 0, kgrp);
  v16bf b2_n = ldfragB(V2, 0, kgrp), b3_n = ldfragB(V3, 0, kgrp);
  for (int kk = 0; kk < kT; kk += 32) {
    v16bf a = a_n, b0 = b0_n, b1 = b1_n, b2 = b2_n, b3 = b3_n;
    const int nk = kk + 32;
    if (nk < kT) {
      a_n  = cvtfragA(arow, nk, kgrp);
      b0_n = ldfragB(V0, nk, kgrp); b1_n = ldfragB(V1, nk, kgrp);
      b2_n = ldfragB(V2, nk, kgrp); b3_n = ldfragB(V3, nk, kgrp);
    }
    c0 = wmma_bf16(a, b0, c0);
    c1 = wmma_bf16(a, b1, c1);
    c2 = wmma_bf16(a, b2, c2);
    c3 = wmma_bf16(a, b3, c3);
  }

  v8f cc[4] = {c0, c1, c2, c3};
#pragma unroll
  for (int j = 0; j < 4; ++j)
#pragma unroll
    for (int i = 0; i < 8; ++i)
      Ctx[((size_t)bh * kT + qtile * 16 + i + 8 * kgrp) * kDK + j * 16 + m] =
          (bf16)cc[j][i];
}

// ============================================================
// Kernel 6: out = Ctx(reshaped b,t,512) @ w_o^T -> f32.
// ============================================================
__global__ __launch_bounds__(32)
void k_outproj(const bf16* __restrict__ Ctx, const bf16* __restrict__ wo,
               float* __restrict__ out) {
  const int tile = blockIdx.x, nt = blockIdx.y;
  const int lane = threadIdx.x;
  const int m = lane & 15, kgrp = lane >> 4;

  const int bt0  = tile * 16;
  const int bIdx = bt0 / kT;
  const int t    = (bt0 % kT) + m;

  const bf16* W0 = wo + (size_t)(nt * 64 +  0 + m) * kDM;
  const bf16* W1 = wo + (size_t)(nt * 64 + 16 + m) * kDM;
  const bf16* W2 = wo + (size_t)(nt * 64 + 32 + m) * kDM;
  const bf16* W3 = wo + (size_t)(nt * 64 + 48 + m) * kDM;

  auto loadActx = [&](int kk) {
    const int hh = kk >> 6;                    // 32-chunks never cross a head
    const int d0 = kk & 63;
    const bf16* rp = Ctx + (((size_t)(bIdx * kH + hh)) * kT + t) * kDK;
    return ldfragA(rp, d0, kgrp);
  };

  v8f c0 = {}, c1 = {}, c2 = {}, c3 = {};
  v16bf a_n  = loadActx(0);
  v16bf b0_n = ldfragB(W0, 0, kgrp), b1_n = ldfragB(W1, 0, kgrp);
  v16bf b2_n = ldfragB(W2, 0, kgrp), b3_n = ldfragB(W3, 0, kgrp);
  for (int kk = 0; kk < kDM; kk += 32) {
    v16bf a = a_n, b0 = b0_n, b1 = b1_n, b2 = b2_n, b3 = b3_n;
    const int nk = kk + 32;
    if (nk < kDM) {
      a_n  = loadActx(nk);
      b0_n = ldfragB(W0, nk, kgrp); b1_n = ldfragB(W1, nk, kgrp);
      b2_n = ldfragB(W2, nk, kgrp); b3_n = ldfragB(W3, nk, kgrp);
    }
    c0 = wmma_bf16(a, b0, c0);
    c1 = wmma_bf16(a, b1, c1);
    c2 = wmma_bf16(a, b2, c2);
    c3 = wmma_bf16(a, b3, c3);
  }

  v8f cc[4] = {c0, c1, c2, c3};
#pragma unroll
  for (int j = 0; j < 4; ++j)
#pragma unroll
    for (int i = 0; i < 8; ++i)
      out[(size_t)(bt0 + i + 8 * kgrp) * kDM + nt * 64 + j * 16 + m] = cc[j][i];
}

// ============================================================
// launch
// ============================================================
extern "C" void kernel_launch(void* const* d_in, const int* in_sizes, int n_in,
                              void* d_out, int out_size, void* d_ws, size_t ws_size,
                              hipStream_t stream) {
  const float* q    = (const float*)d_in[0];
  const float* k    = (const float*)d_in[1];
  const float* v    = (const float*)d_in[2];
  const int*   mask = (const int*)  d_in[3];
  const float* prev = (const float*)d_in[4];
  const float* w_q  = (const float*)d_in[5];
  const float* w_k  = (const float*)d_in[6];
  const float* w_v  = (const float*)d_in[7];
  const float* w_o  = (const float*)d_in[8];
  const float* tw   = (const float*)d_in[9];
  const float* tb   = (const float*)d_in[10];
  const float* aw   = (const float*)d_in[11];
  const float* ab   = (const float*)d_in[12];

  constexpr size_t nX   = (size_t)kB * kT * kDM;   // 2,097,152
  constexpr size_t nW   = (size_t)kDM * kDM;       //   262,144
  constexpr size_t nP   = (size_t)kB * kH * kT * kDK;
  constexpr size_t nMap = (size_t)kB * kH * kT * kT;

  char* p = (char*)d_ws;
  bf16* qb   = (bf16*)p; p += nX * 2;
  bf16* kb   = (bf16*)p; p += nX * 2;
  bf16* vb   = (bf16*)p; p += nX * 2;
  bf16* wqb  = (bf16*)p; p += nW * 2;
  bf16* wkb  = (bf16*)p; p += nW * 2;
  bf16* wvb  = (bf16*)p; p += nW * 2;
  bf16* wob  = (bf16*)p; p += nW * 2;
  bf16* Qw   = (bf16*)p; p += nP * 2;
  bf16* Kw   = (bf16*)p; p += nP * 2;
  bf16* Vtw  = (bf16*)p; p += nP * 2;
  bf16* Ctxw = (bf16*)p; p += nP * 2;
  bf16* Mw   = (bf16*)p; p += nMap * 2;
  bf16* Mtw  = (bf16*)p; p += nMap * 2;

  float* out = (float*)d_out;                  // (B,T,512)
  float* A   = out + (size_t)kB * kT * kDM;    // (B,H,T,T)

  // 0. bf16 conversions
  k_cvt<<<2048, 256, 0, stream>>>(q, qb, (int)nX);
  k_cvt<<<2048, 256, 0, stream>>>(k, kb, (int)nX);
  k_cvt<<<2048, 256, 0, stream>>>(v, vb, (int)nX);
  k_cvt<<<512, 256, 0, stream>>>(w_q, wqb, (int)nW);
  k_cvt<<<512, 256, 0, stream>>>(w_k, wkb, (int)nW);
  k_cvt<<<512, 256, 0, stream>>>(w_v, wvb, (int)nW);
  k_cvt<<<512, 256, 0, stream>>>(w_o, wob, (int)nW);
  // 1. QKV projections (bf16 WMMA)
  k_qkv<<<dim3(kB * kT / 16, kH, 3), 32, 0, stream>>>(qb, kb, vb, wqb, wkb, wvb,
                                                      Qw, Kw, Vtw);
  // 2. M = Q K^T (bf16 WMMA, bf16 out)
  k_scores<<<dim3(kT / 64, kT / 16, kB * kH), 32, 0, stream>>>(Qw, Kw, Mw);
  // 3. Mt = conv(prev, tw) + tb (TDM-staged, VALU, bf16 out)
  k_tconv<<<dim3(kT, kB), 256, kTconvLds, stream>>>(prev, tw, tb, Mtw);
  // 4. A = softmax(mask(conv(concat(M,Mt), aw)/8)) (TDM-staged)
  k_aconv_softmax<<<dim3(kT, kB), 256, kAconvLds, stream>>>(Mw, Mtw, aw, ab,
                                                            mask, A);
  // 5. Ctx = A V (bf16 WMMA)
  k_context<<<dim3(kT / 16, kB * kH), 32, 0, stream>>>(A, Vtw, Ctxw);
  // 6. out = Ctx @ w_o^T (bf16 WMMA)
  k_outproj<<<dim3(kB * kT / 16, kDM / 64), 32, 0, stream>>>(Ctxw, wob, out);
}
